// KWattention_45226005627030
// MI455X (gfx1250) — compile-verified, matrix operand order checked by reading, not run on hardware
//
#include <hip/hip_runtime.h>
#include <hip/hip_bf16.h>
#include <math.h>

typedef __attribute__((ext_vector_type(16))) __bf16 v16bf;
typedef __attribute__((ext_vector_type(8)))  __bf16 v8bf;
typedef __attribute__((ext_vector_type(8)))  float  v8f;

// ---------------------------------------------------------------------------
// Fragment loader for 16x32 bf16 A/B operands of v_wmma_f32_16x16x32_bf16.
// Per CDNA5 ISA 7.12.2 (16-bit A-matrix 16x32): lane L holds row M=L%16,
// half = L/16 selects K groups: regs 0..3 -> K = half*8 + {0..7},
// regs 4..7 -> K = 16 + half*8 + {0..7}. Two contiguous 16B loads per lane.
// B operand is supplied from a K-major (transposed) array so the same loader
// applies with lane indexing the N dimension.
// ---------------------------------------------------------------------------
__device__ __forceinline__ v16bf load_frag_rowK(const __bf16* base, int ld) {
  const int lane = threadIdx.x & 31;
  const int r    = lane & 15;
  const int half = lane >> 4;
  const __bf16* p = base + (size_t)r * ld + half * 8;
  v8bf lo = *reinterpret_cast<const v8bf*>(p);
  v8bf hi = *reinterpret_cast<const v8bf*>(p + 16);
  v16bf f;
#pragma unroll
  for (int j = 0; j < 8; ++j) { f[j] = lo[j]; f[j + 8] = hi[j]; }
  return f;
}

// ---------------------------------------------------------------------------
// Generic bf16 WMMA GEMM: C[M,N] = A[M,K] * Bt[N,K]^T  (fp32 accumulate)
// 128 threads = 4 waves stacked along M; each wave computes a 16x64 strip
// (A fragment loaded once per K-step, reused across 4 B tiles -> 2.5 b128
// loads per WMMA instead of 4). grid = (N/64, M/64).
// mode 0: outF = acc
// mode 1: outB = bf16(gelu(acc + bias[n]))
// mode 2: outF = acc + bias[n] + res[m,n]
// mode 3: x = acc + bias[n]; outF = x; outB = bf16(x)
// ---------------------------------------------------------------------------
__global__ __launch_bounds__(128) void gemm_bf16_k(
    const __bf16* __restrict__ A, const __bf16* __restrict__ Bt,
    int M, int N, int K,
    const float* __restrict__ bias, const float* __restrict__ res,
    float* __restrict__ outF, __bf16* __restrict__ outB, int mode)
{
  const int wv   = threadIdx.x >> 5;
  const int lane = threadIdx.x & 31;
  const int m0 = (blockIdx.y * 4 + wv) * 16;
  const int n0 = blockIdx.x * 64;
  if (m0 >= M || n0 >= N) return;

  v8f acc[4];
#pragma unroll
  for (int t = 0; t < 4; ++t) acc[t] = (v8f){0.f,0.f,0.f,0.f,0.f,0.f,0.f,0.f};

  const __bf16* a0 = A  + (size_t)m0 * K;
  const __bf16* b0 = Bt + (size_t)n0 * K;
  for (int k0 = 0; k0 < K; k0 += 32) {
    v16bf a = load_frag_rowK(a0 + k0, K);
#pragma unroll
    for (int nt = 0; nt < 4; ++nt) {
      v16bf b = load_frag_rowK(b0 + (size_t)nt * 16 * K + k0, K);
      acc[nt] = __builtin_amdgcn_wmma_f32_16x16x32_bf16(false, a, false, b,
                                                        (short)0, acc[nt], false, false);
    }
  }
  // D layout: reg v, lane -> m = v + 8*(lane/16), n = lane%16
  const int nn = lane & 15;
  const int mh = (lane >> 4) * 8;
#pragma unroll
  for (int nt = 0; nt < 4; ++nt) {
#pragma unroll
    for (int v = 0; v < 8; ++v) {
      const int m = m0 + mh + v;
      const int n = n0 + nt * 16 + nn;
      const size_t idx = (size_t)m * N + n;
      float x = acc[nt][v];
      if (mode == 0) {
        outF[idx] = x;
      } else if (mode == 1) {
        x += bias[n];
        float g = 0.5f * x * (1.0f + erff(x * 0.70710678118654752f));
        outB[idx] = (__bf16)g;
      } else if (mode == 2) {
        x += bias[n] + res[idx];
        outF[idx] = x;
      } else {
        x += bias[n];
        outF[idx] = x;
        outB[idx] = (__bf16)x;
      }
    }
  }
}

// ---------------------------------------------------------------------------
// Keyword attention. Block = (b, h, chunk of 10 keywords). 128 thr = 4 waves.
// For each keyword: Q/K/V = P{Q,K,V}(head slice) * kw (f32 -> bf16 in LDS),
// S = Q K^T * scale + mask, softmax over the QUERY axis (axis=-2, per ref),
// ctx += softmax(S) @ V, all contractions via v_wmma_f32_16x16x32_bf16.
// Accumulate Wmlp[kw]*ctx in registers; one atomicAdd scatter per block.
// ---------------------------------------------------------------------------
__global__ __launch_bounds__(128) void attn_k(
    const float* __restrict__ PQ, const float* __restrict__ PK,
    const float* __restrict__ PV, const float* __restrict__ pool,
    const float* __restrict__ amask, const float* __restrict__ Wmlp,
    float* __restrict__ ctxsum)
{
  const int CH = 10;                 // keywords per block
  const int bi = blockIdx.x;
  const int kc = bi % 10;            // keyword chunk
  const int h  = (bi / 10) % 12;     // head
  const int b  = bi / 120;           // batch
  const int tid  = threadIdx.x;
  const int wv   = tid >> 5;
  const int lane = tid & 31;

  __shared__ float kwv[64];
  __shared__ __align__(16) __bf16 Qb[128 * 64];   // 16 KB
  __shared__ __align__(16) __bf16 Kb[32 * 64];    //  4 KB
  __shared__ __align__(16) __bf16 VbT[64 * 32];   //  4 KB (V transposed: [d][k'])
  __shared__ float Sc[128 * 32];                  // 16 KB
  __shared__ float red[128];                      // softmax partials
  __shared__ float cmax[32], cinv[32];
  __shared__ __align__(16) __bf16 Wc[128 * 32];   //  8 KB

  const float scale = 0.125f;        // 1/sqrt(64)
  const size_t base = ((size_t)b * 128) * 768 + (size_t)h * 64;

  v8f tot[8];
#pragma unroll
  for (int t = 0; t < 8; ++t) tot[t] = (v8f){0.f,0.f,0.f,0.f,0.f,0.f,0.f,0.f};

  for (int kk = 0; kk < CH; ++kk) {
    const int kw = kc * CH + kk;
    if (tid < 64) kwv[tid] = pool[(size_t)kw * 768 + (size_t)h * 64 + tid];
    const float wml = Wmlp[kw];
    __syncthreads();

    // stage modulated Q: [128 s][64 d] bf16
    for (int i = tid; i < 128 * 64; i += 128) {
      const int s = i >> 6, d = i & 63;
      Qb[i] = (__bf16)(PQ[base + (size_t)s * 768 + d] * kwv[d]);
    }

    v8f ctx[8];
#pragma unroll
    for (int t = 0; t < 8; ++t) ctx[t] = (v8f){0.f,0.f,0.f,0.f,0.f,0.f,0.f,0.f};

    for (int c = 0; c < 4; ++c) {        // 32-wide key-column chunks
      __syncthreads();                    // Qb ready / Kb,VbT,Sc reusable
      for (int i = tid; i < 32 * 64; i += 128) {
        const int r = i >> 6, d = i & 63;
        const int s = c * 32 + r;
        const float kv = kwv[d];
        Kb[i]           = (__bf16)(PK[base + (size_t)s * 768 + d] * kv);
        VbT[d * 32 + r] = (__bf16)(PV[base + (size_t)s * 768 + d] * kv);
      }
      __syncthreads();

      // scores: wave wv owns q-tiles {2wv,2wv+1} x k'-tiles {0,1}
#pragma unroll
      for (int qt = 0; qt < 2; ++qt) {
        const int qi = wv * 2 + qt;
#pragma unroll
        for (int ct = 0; ct < 2; ++ct) {
          v8f s8 = {0.f,0.f,0.f,0.f,0.f,0.f,0.f,0.f};
#pragma unroll
          for (int k0 = 0; k0 < 64; k0 += 32) {
            v16bf a  = load_frag_rowK(Qb + qi * 16 * 64 + k0, 64);
            v16bf bb = load_frag_rowK(Kb + ct * 16 * 64 + k0, 64);
            s8 = __builtin_amdgcn_wmma_f32_16x16x32_bf16(false, a, false, bb,
                                                         (short)0, s8, false, false);
          }
          const int nn = lane & 15, mh = (lane >> 4) * 8;
#pragma unroll
          for (int v = 0; v < 8; ++v) {
            const int q = qi * 16 + mh + v;
            const int n = ct * 16 + nn;
            Sc[q * 32 + n] = s8[v] * scale + amask[b * 128 + c * 32 + n];
          }
        }
      }
      __syncthreads();

      // column softmax (over q axis = axis -2 of the reference), 4-way parallel
      {
        const int col = tid & 31, seg = tid >> 5;   // 4 segments of 32 rows
        float mx = -3.4e38f;
        for (int q = seg * 32; q < seg * 32 + 32; ++q)
          mx = fmaxf(mx, Sc[q * 32 + col]);
        red[seg * 32 + col] = mx;
      }
      __syncthreads();
      if (tid < 32)
        cmax[tid] = fmaxf(fmaxf(red[tid], red[32 + tid]),
                          fmaxf(red[64 + tid], red[96 + tid]));
      __syncthreads();
      {
        const int col = tid & 31, seg = tid >> 5;
        const float mx = cmax[col];
        float sm = 0.f;
        for (int q = seg * 32; q < seg * 32 + 32; ++q)
          sm += __expf(Sc[q * 32 + col] - mx);
        red[seg * 32 + col] = sm;
      }
      __syncthreads();
      if (tid < 32)
        cinv[tid] = 1.f / (red[tid] + red[32 + tid] + red[64 + tid] + red[96 + tid]);
      __syncthreads();
      for (int i = tid; i < 128 * 32; i += 128) {
        const int n = i & 31;
        Wc[i] = (__bf16)(__expf(Sc[i] - cmax[n]) * cinv[n]);
      }
      __syncthreads();

      // ctx += Wc[128x32] @ V[32x64] : one K=32 WMMA step per tile
#pragma unroll
      for (int qt = 0; qt < 2; ++qt) {
        const int qi = wv * 2 + qt;
        v16bf a = load_frag_rowK(Wc + qi * 16 * 32, 32);
#pragma unroll
        for (int dj = 0; dj < 4; ++dj) {
          v16bf bb = load_frag_rowK(VbT + dj * 16 * 32, 32);
          ctx[qt * 4 + dj] = __builtin_amdgcn_wmma_f32_16x16x32_bf16(
              false, a, false, bb, (short)0, ctx[qt * 4 + dj], false, false);
        }
      }
    }

#pragma unroll
    for (int t = 0; t < 8; ++t)
#pragma unroll
      for (int v = 0; v < 8; ++v) tot[t][v] += wml * ctx[t][v];
    __syncthreads();                     // guard LDS reuse before next keyword
  }

  // scatter Σ Wmlp[k]*ctx_k into ctxsum (head-concatenated [B*S, H] layout)
  const int nn = lane & 15, mh = (lane >> 4) * 8;
#pragma unroll
  for (int qt = 0; qt < 2; ++qt) {
    const int qi = wv * 2 + qt;
#pragma unroll
    for (int dj = 0; dj < 4; ++dj) {
#pragma unroll
      for (int v = 0; v < 8; ++v) {
        const int q = qi * 16 + mh + v;
        const int d = dj * 16 + nn;
        atomicAdd(&ctxsum[base + (size_t)q * 768 + d], tot[qt * 4 + dj][v]);
      }
    }
  }
}

// ------------------------------ prep / epilogue ----------------------------
__global__ void pack_pool_k(const float* __restrict__ pos, const float* __restrict__ neg,
                            float* __restrict__ pool, int P, int H) {
  int i = blockIdx.x * blockDim.x + threadIdx.x;
  if (i >= 2 * P * H) return;
  int r = i / H, j = i % H;
  pool[i] = (r & 1) ? neg[(r >> 1) * H + j] : pos[(r >> 1) * H + j];
}

__global__ void f2b_k(const float* __restrict__ s, __bf16* __restrict__ d, int n) {
  int i = blockIdx.x * blockDim.x + threadIdx.x;
  if (i < n) d[i] = (__bf16)s[i];
}

__global__ void transpose_f2b_k(const float* __restrict__ s, __bf16* __restrict__ d,
                                int K, int N) {
  int i = blockIdx.x * blockDim.x + threadIdx.x;
  if (i >= K * N) return;
  int k = i / N, n = i % N;
  d[(size_t)n * K + k] = (__bf16)s[i];
}

__global__ void zero_k(float* __restrict__ p, int n) {
  int i = blockIdx.x * blockDim.x + threadIdx.x;
  if (i < n) p[i] = 0.f;
}

__global__ void cbias_k(const float* __restrict__ Wmlp, const float* __restrict__ bo,
                        const float* __restrict__ bmlp, float* __restrict__ cbias,
                        int KW, int H) {
  __shared__ float red[256];
  int tid = threadIdx.x;
  float s = 0.f;
  for (int i = tid; i < KW; i += 256) s += Wmlp[i];
  red[tid] = s; __syncthreads();
  for (int st = 128; st > 0; st >>= 1) {
    if (tid < st) red[tid] += red[tid + st];
    __syncthreads();
  }
  float ws = red[0];
  for (int n = tid; n < H; n += 256) cbias[n] = bo[n] * ws + bmlp[0];
}

__global__ void ln_k(const float* __restrict__ x, const float* __restrict__ gamma,
                     const float* __restrict__ beta, float* __restrict__ out, int H) {
  __shared__ float s1[256], s2[256];
  int row = blockIdx.x, tid = threadIdx.x;
  const float* xr = x + (size_t)row * H;
  float a = 0.f, b2 = 0.f;
  for (int j = tid; j < H; j += 256) { float v = xr[j]; a += v; b2 += v * v; }
  s1[tid] = a; s2[tid] = b2; __syncthreads();
  for (int st = 128; st > 0; st >>= 1) {
    if (tid < st) { s1[tid] += s1[tid + st]; s2[tid] += s2[tid + st]; }
    __syncthreads();
  }
  float mu  = s1[0] / H;
  float var = s2[0] / H - mu * mu;
  float rs  = rsqrtf(var + 1e-12f);
  float* orow = out + (size_t)row * H;
  for (int j = tid; j < H; j += 256) orow[j] = gamma[j] * (xr[j] - mu) * rs + beta[j];
}

// ---------------------------------------------------------------------------
extern "C" void kernel_launch(void* const* d_in, const int* in_sizes, int n_in,
                              void* d_out, int out_size, void* d_ws, size_t ws_size,
                              hipStream_t stream) {
  const float* hidden = (const float*)d_in[0];
  const float* pos    = (const float*)d_in[1];
  const float* neg    = (const float*)d_in[2];
  const float* amask  = (const float*)d_in[3];
  const float* Wq     = (const float*)d_in[4];
  const float* Wk     = (const float*)d_in[5];
  const float* Wv     = (const float*)d_in[6];
  const float* Wo     = (const float*)d_in[7];
  const float* bo     = (const float*)d_in[8];
  const float* Wmlp   = (const float*)d_in[9];
  const float* bmlp   = (const float*)d_in[10];
  const float* Wdown  = (const float*)d_in[11];
  const float* bdown  = (const float*)d_in[12];
  const float* Wup    = (const float*)d_in[13];
  const float* bup    = (const float*)d_in[14];
  const float* gamma  = (const float*)d_in[15];
  const float* beta   = (const float*)d_in[16];
  float* out = (float*)d_out;

  const int B = 8, S = 128, H = 768, I = 3072, KW = 100, P = 50;
  const int M = B * S;   // 1024

  // bump allocator over workspace (recomputed identically every call)
  char* w = (char*)d_ws;
  auto alloc = [&](size_t bytes) -> void* {
    void* p = (void*)w;
    w += (bytes + 255) & ~(size_t)255;
    return p;
  };
  float*  pool   = (float*) alloc((size_t)KW * H * 4);
  __bf16* hidB   = (__bf16*)alloc((size_t)M * H * 2);
  __bf16* WqT    = (__bf16*)alloc((size_t)H * H * 2);
  __bf16* WkT    = (__bf16*)alloc((size_t)H * H * 2);
  __bf16* WvT    = (__bf16*)alloc((size_t)H * H * 2);
  __bf16* WoT    = (__bf16*)alloc((size_t)H * H * 2);
  __bf16* WdownT = (__bf16*)alloc((size_t)H * I * 2);
  __bf16* WupT   = (__bf16*)alloc((size_t)I * H * 2);
  float*  PQ     = (float*) alloc((size_t)M * H * 4);
  float*  PK     = (float*) alloc((size_t)M * H * 4);
  float*  PV     = (float*) alloc((size_t)M * H * 4);
  float*  ctxsum = (float*) alloc((size_t)M * H * 4);
  __bf16* csB    = (__bf16*)alloc((size_t)M * H * 2);
  float*  fusedF = (float*) alloc((size_t)M * H * 4);
  __bf16* fusedB = (__bf16*)alloc((size_t)M * H * 2);
  __bf16* downB  = (__bf16*)alloc((size_t)M * I * 2);
  float*  xbuf   = (float*) alloc((size_t)M * H * 4);
  float*  cbias  = (float*) alloc((size_t)H * 4);

  auto g1 = [](int n) { return dim3((n + 255) / 256); };

  // prep
  pack_pool_k<<<g1(KW * H), 256, 0, stream>>>(pos, neg, pool, P, H);
  f2b_k<<<g1(M * H), 256, 0, stream>>>(hidden, hidB, M * H);
  transpose_f2b_k<<<g1(H * H), 256, 0, stream>>>(Wq, WqT, H, H);
  transpose_f2b_k<<<g1(H * H), 256, 0, stream>>>(Wk, WkT, H, H);
  transpose_f2b_k<<<g1(H * H), 256, 0, stream>>>(Wv, WvT, H, H);
  transpose_f2b_k<<<g1(H * H), 256, 0, stream>>>(Wo, WoT, H, H);
  transpose_f2b_k<<<g1(H * I), 256, 0, stream>>>(Wdown, WdownT, H, I);
  transpose_f2b_k<<<g1(I * H), 256, 0, stream>>>(Wup, WupT, I, H);
  zero_k<<<g1(M * H), 256, 0, stream>>>(ctxsum, M * H);
  cbias_k<<<1, 256, 0, stream>>>(Wmlp, bo, bmlp, cbias, KW, H);

  // QKV projections: [1024,768] = hidB @ W{q,k,v}
  dim3 gHH(H / 64, M / 64);
  gemm_bf16_k<<<gHH, 128, 0, stream>>>(hidB, WqT, M, H, H, nullptr, nullptr, PQ, nullptr, 0);
  gemm_bf16_k<<<gHH, 128, 0, stream>>>(hidB, WkT, M, H, H, nullptr, nullptr, PK, nullptr, 0);
  gemm_bf16_k<<<gHH, 128, 0, stream>>>(hidB, WvT, M, H, H, nullptr, nullptr, PV, nullptr, 0);

  // keyword attention -> ctxsum = Σ Wmlp[k] * ctx_k
  attn_k<<<B * 12 * 10, 128, 0, stream>>>(PQ, PK, PV, pool, amask, Wmlp, ctxsum);

  // fused = ctxsum @ Wo + (bo*ΣWmlp + bmlp)
  f2b_k<<<g1(M * H), 256, 0, stream>>>(ctxsum, csB, M * H);
  gemm_bf16_k<<<gHH, 128, 0, stream>>>(csB, WoT, M, H, H, cbias, nullptr, fusedF, fusedB, 3);

  // FFN: down = gelu(fused @ Wdown + bdown); up = down @ Wup + bup; x = up + fused
  dim3 gHI(I / 64, M / 64);
  gemm_bf16_k<<<gHI, 128, 0, stream>>>(fusedB, WdownT, M, I, H, bdown, nullptr, nullptr, downB, 1);
  gemm_bf16_k<<<gHH, 128, 0, stream>>>(downB, WupT, M, H, I, bup, fusedF, xbuf, nullptr, 2);

  // LayerNorm -> output
  ln_k<<<M, 256, 0, stream>>>(xbuf, gamma, beta, out, H);
}
